// MiniMaxForCausalLM_59803124630223
// MI455X (gfx1250) — matrix operand order repaired and checked by reading, not compiled, not run on hardware
//
#include <hip/hip_runtime.h>
#include <hip/hip_bf16.h>

typedef __attribute__((ext_vector_type(16))) __bf16 v16bf;
typedef __attribute__((ext_vector_type(8)))  __bf16 v8bf;
typedef __attribute__((ext_vector_type(8)))  float  v8f;

namespace {
constexpr int kT = 8192;   // B*S tokens
constexpr int kH = 2048;
constexpr int kF = 1408;
constexpr int kE = 8;
constexpr int BM = 128;    // token tile
constexpr int BN = 64;     // output-feature tile
constexpr int BK = 32;     // K step (bf16 WMMA K=32)
}

// ---------------------------------------------------------------------------
// Router: logits = x @ w_router, top-2, normalized softmax weights -> coef[T,E]
// One block (128 threads = 4 waves) per token.
// ---------------------------------------------------------------------------
__global__ __launch_bounds__(128)
void router_topk_kernel(const float* __restrict__ x,
                        const float* __restrict__ wr,
                        float* __restrict__ coef) {
  const int t   = blockIdx.x;
  const int tid = threadIdx.x;
  float acc[kE];
#pragma unroll
  for (int e = 0; e < kE; ++e) acc[e] = 0.f;

  const float* xr = x + (size_t)t * kH;
  for (int h = tid; h < kH; h += 128) {
    const float xv = xr[h];
    const float* w = wr + (size_t)h * kE;
#pragma unroll
    for (int e = 0; e < kE; ++e) acc[e] += xv * w[e];
  }
  // wave32 reduction
#pragma unroll
  for (int e = 0; e < kE; ++e) {
#pragma unroll
    for (int off = 16; off > 0; off >>= 1) acc[e] += __shfl_xor(acc[e], off, 32);
  }
  __shared__ float sred[4][kE];
  const int wv = tid >> 5, ln = tid & 31;
  if (ln == 0) {
#pragma unroll
    for (int e = 0; e < kE; ++e) sred[wv][e] = acc[e];
  }
  __syncthreads();
  if (tid == 0) {
    float l[kE];
#pragma unroll
    for (int e = 0; e < kE; ++e)
      l[e] = sred[0][e] + sred[1][e] + sred[2][e] + sred[3][e];
    int i1 = 0;
#pragma unroll
    for (int e = 1; e < kE; ++e) if (l[e] > l[i1]) i1 = e;
    int i2 = (i1 == 0) ? 1 : 0;
#pragma unroll
    for (int e = 0; e < kE; ++e) if (e != i1 && l[e] > l[i2]) i2 = e;
    // normalized top-2 softmax weights == 2-way softmax of the two top logits
    const float w1 = 1.f / (1.f + __expf(l[i2] - l[i1]));
    const float w2 = 1.f - w1;
    float* c = coef + (size_t)t * kE;
#pragma unroll
    for (int e = 0; e < kE; ++e) c[e] = (e == i1) ? w1 : ((e == i2) ? w2 : 0.f);
  }
}

// ---------------------------------------------------------------------------
// Fragment helpers (ISA 7.12.2 wave32 layouts for 16x16x32 bf16 WMMA)
// ---------------------------------------------------------------------------
__device__ __forceinline__ v16bf load_a_frag(const __bf16* As, int mbase,
                                             int r15, int hsel) {
  // A 16x32: lanes 0-15 row M=lane, K in {0..7, 16..23}; lanes 16-31 K in {8..15, 24..31}
  const int row = mbase + r15;
  const int kb  = hsel * 8;
  const v8bf lo = *(const v8bf*)&As[row * BK + kb];
  const v8bf hi = *(const v8bf*)&As[row * BK + kb + 16];
  v16bf a;
#pragma unroll
  for (int j = 0; j < 8; ++j) { a[j] = lo[j]; a[8 + j] = hi[j]; }
  return a;
}

__device__ __forceinline__ v16bf load_b_frag(const __bf16* BT, int nbase,
                                             int r15, int hsel) {
  // B 32x16: lane = column; lanes 0-15 hold K in {0..7, 16..23}, lanes 16-31 K in {8..15, 24..31}.
  // BT is stored [n][k] so each lane reads two runs of 8 contiguous bf16 (2 x ds_load_b128).
  const int col = nbase + r15;
  const int kb  = hsel * 8;
  const v8bf lo = *(const v8bf*)&BT[col * BK + kb];
  const v8bf hi = *(const v8bf*)&BT[col * BK + kb + 16];
  v16bf b;
#pragma unroll
  for (int j = 0; j < 8; ++j) { b[j] = lo[j]; b[8 + j] = hi[j]; }
  return b;
}

// Async global->LDS copy of 16 bytes (CDNA5, tracked with ASYNCcnt).
__device__ __forceinline__ void async_copy_b128(__bf16* lds_dst, const __bf16* gsrc) {
  // Generic shared pointer: addr[31:0] is the LDS offset (ISA 10.2 aperture rules).
  const unsigned ldsoff = (unsigned)(unsigned long long)(void*)lds_dst;
  asm volatile("global_load_async_to_lds_b128 %0, %1, off"
               :
               : "v"(ldsoff), "v"(gsrc)
               : "memory");
}
__device__ __forceinline__ void wait_asynccnt0() {
  asm volatile("s_wait_asynccnt 0x0" ::: "memory");
}

// ---------------------------------------------------------------------------
// Phase 1 per expert: hid[T,F] = silu(x@Wg) * (x@Wu)  (bf16 output)
// Block tile 128x64, 8 waves (4M x 2N), each wave 32x32 per matrix.
// ---------------------------------------------------------------------------
__global__ __launch_bounds__(256)
void moe_gateup_wmma(const float* __restrict__ x,
                     const float* __restrict__ wg,
                     const float* __restrict__ wu,
                     const float* __restrict__ coef, int eidx,
                     __bf16* __restrict__ hid) {
  __shared__ __align__(16) __bf16 As[BM * BK];
  __shared__ __align__(16) __bf16 BgT[BN * BK];
  __shared__ __align__(16) __bf16 BuT[BN * BK];
  __shared__ int s_any;

  const int tid  = threadIdx.x;
  const int lane = tid & 31;
  const int wv   = tid >> 5;
  const int wm   = wv & 3;        // 0..3 -> M quadrant
  const int wn   = wv >> 2;       // 0..1 -> N half
  const int t0   = blockIdx.y * BM;
  const int f0   = blockIdx.x * BN;
  const int r15  = lane & 15;
  const int hsel = lane >> 4;

  // block-sparse routing: skip whole token tile if no token hits this expert
  if (tid == 0) s_any = 0;
  __syncthreads();
  if (tid < BM && coef[(size_t)(t0 + tid) * kE + eidx] != 0.f) atomicOr(&s_any, 1);
  __syncthreads();
  if (!s_any) return;

  const v8f vz = {0.f, 0.f, 0.f, 0.f, 0.f, 0.f, 0.f, 0.f};
  v8f accG[2][2], accU[2][2];
#pragma unroll
  for (int mi = 0; mi < 2; ++mi)
#pragma unroll
    for (int ni = 0; ni < 2; ++ni) { accG[mi][ni] = vz; accU[mi][ni] = vz; }

  for (int k0 = 0; k0 < kH; k0 += BK) {
    // stage x tile (fp32 -> bf16), [t][k]
    for (int i = tid; i < BM * BK / 4; i += 256) {
      const int row = i >> 3;
      const int c   = (i & 7) << 2;
      const float4 v = *(const float4*)(x + (size_t)(t0 + row) * kH + k0 + c);
      __bf16* d = &As[row * BK + c];
      d[0] = (__bf16)v.x; d[1] = (__bf16)v.y; d[2] = (__bf16)v.z; d[3] = (__bf16)v.w;
    }
    // stage Wg/Wu tiles (fp32 -> bf16), K-transposed: [f][k]
    for (int i = tid; i < BK * BN / 4; i += 256) {
      const int row = i >> 4;          // k 0..31
      const int c   = (i & 15) << 2;   // f 0..60
      const size_t g = (size_t)(k0 + row) * kF + f0 + c;
      const float4 vg = *(const float4*)(wg + g);
      const float4 vu = *(const float4*)(wu + g);
      BgT[(c + 0) * BK + row] = (__bf16)vg.x;
      BgT[(c + 1) * BK + row] = (__bf16)vg.y;
      BgT[(c + 2) * BK + row] = (__bf16)vg.z;
      BgT[(c + 3) * BK + row] = (__bf16)vg.w;
      BuT[(c + 0) * BK + row] = (__bf16)vu.x;
      BuT[(c + 1) * BK + row] = (__bf16)vu.y;
      BuT[(c + 2) * BK + row] = (__bf16)vu.z;
      BuT[(c + 3) * BK + row] = (__bf16)vu.w;
    }
    if (k0 + BK < kH) {
      __builtin_prefetch(x + (size_t)(t0 + (tid >> 1)) * kH + k0 + BK, 0, 1);
      __builtin_prefetch(wg + (size_t)(k0 + BK + lane) * kF + f0, 0, 1);
      __builtin_prefetch(wu + (size_t)(k0 + BK + lane) * kF + f0, 0, 1);
    }
    __syncthreads();

    v16bf a[2], bg[2], bu[2];
#pragma unroll
    for (int mi = 0; mi < 2; ++mi) a[mi] = load_a_frag(As, wm * 32 + mi * 16, r15, hsel);
#pragma unroll
    for (int ni = 0; ni < 2; ++ni) {
      bg[ni] = load_b_frag(BgT, wn * 32 + ni * 16, r15, hsel);
      bu[ni] = load_b_frag(BuT, wn * 32 + ni * 16, r15, hsel);
    }
#pragma unroll
    for (int mi = 0; mi < 2; ++mi)
#pragma unroll
      for (int ni = 0; ni < 2; ++ni) {
        accG[mi][ni] = __builtin_amdgcn_wmma_f32_16x16x32_bf16(
            false, a[mi], false, bg[ni], (short)0, accG[mi][ni], false, false);
        accU[mi][ni] = __builtin_amdgcn_wmma_f32_16x16x32_bf16(
            false, a[mi], false, bu[ni], (short)0, accU[mi][ni], false, false);
      }
    __syncthreads();
  }

  // epilogue: hid = silu(g) * u, bf16 (C layout: VGPR r -> M=r (lanes<16) / r+8)
#pragma unroll
  for (int mi = 0; mi < 2; ++mi)
#pragma unroll
    for (int ni = 0; ni < 2; ++ni) {
      const int trow = t0 + wm * 32 + mi * 16 + hsel * 8;
      const int fcol = f0 + wn * 32 + ni * 16 + r15;
      const v8f g = accG[mi][ni];
      const v8f u = accU[mi][ni];
#pragma unroll
      for (int r = 0; r < 8; ++r) {
        const float gv = g[r];
        const float hv = (gv / (1.f + __expf(-gv))) * u[r];
        hid[(size_t)(trow + r) * kF + fcol] = (__bf16)hv;
      }
    }
}

// ---------------------------------------------------------------------------
// Phase 2 per expert: out[T,H] (+)= coef[:,e] * (hid @ Wd)
// A tile (bf16 hid) is staged with the CDNA5 async DMA path (ASYNCcnt).
// ---------------------------------------------------------------------------
__global__ __launch_bounds__(256)
void moe_down_wmma(const __bf16* __restrict__ hid,
                   const float* __restrict__ wd,
                   const float* __restrict__ coef, int eidx,
                   float* __restrict__ out, int accumulate) {
  __shared__ __align__(16) __bf16 As[BM * BK];
  __shared__ __align__(16) __bf16 BT[BN * BK];
  __shared__ int s_any;

  const int tid  = threadIdx.x;
  const int lane = tid & 31;
  const int wv   = tid >> 5;
  const int wm   = wv & 3;
  const int wn   = wv >> 2;
  const int t0   = blockIdx.y * BM;
  const int h0   = blockIdx.x * BN;
  const int r15  = lane & 15;
  const int hsel = lane >> 4;

  if (tid == 0) s_any = 0;
  __syncthreads();
  if (tid < BM && coef[(size_t)(t0 + tid) * kE + eidx] != 0.f) atomicOr(&s_any, 1);
  __syncthreads();
  if (!s_any) {
    if (!accumulate) {  // expert 0 initializes d_out: write zeros for skipped tile
      for (int i = tid; i < BM * BN; i += 256)
        out[(size_t)(t0 + i / BN) * kH + h0 + (i % BN)] = 0.f;
    }
    return;
  }

  const v8f vz = {0.f, 0.f, 0.f, 0.f, 0.f, 0.f, 0.f, 0.f};
  v8f acc[2][2];
#pragma unroll
  for (int mi = 0; mi < 2; ++mi)
#pragma unroll
    for (int ni = 0; ni < 2; ++ni) acc[mi][ni] = vz;

  for (int k0 = 0; k0 < kF; k0 += BK) {
    // stage hid tile (bf16, no conversion): async global->LDS DMA, 16B/lane.
    // 128x32 bf16 = 8 KB = 512 x b128; 256 threads -> 2 issues each.
#pragma unroll
    for (int i = tid; i < BM * BK / 8; i += 256) {
      const int row = i >> 2;
      const int c   = (i & 3) << 3;
      async_copy_b128(&As[row * BK + c],
                      hid + (size_t)(t0 + row) * kF + k0 + c);
    }
    // stage Wd tile (fp32 -> bf16), K-transposed [h][k]
    for (int i = tid; i < BK * BN / 4; i += 256) {
      const int row = i >> 4;
      const int c   = (i & 15) << 2;
      const float4 v = *(const float4*)(wd + (size_t)(k0 + row) * kH + h0 + c);
      BT[(c + 0) * BK + row] = (__bf16)v.x;
      BT[(c + 1) * BK + row] = (__bf16)v.y;
      BT[(c + 2) * BK + row] = (__bf16)v.z;
      BT[(c + 3) * BK + row] = (__bf16)v.w;
    }
    if (k0 + BK < kF) {
      __builtin_prefetch(hid + (size_t)(t0 + (tid >> 1)) * kF + k0 + BK, 0, 1);
      __builtin_prefetch(wd + (size_t)(k0 + BK + lane) * kH + h0, 0, 1);
    }
    wait_asynccnt0();   // all async LDS writes from this wave complete
    __syncthreads();    // ... and from all waves in the group

    v16bf a[2], b[2];
#pragma unroll
    for (int mi = 0; mi < 2; ++mi) a[mi] = load_a_frag(As, wm * 32 + mi * 16, r15, hsel);
#pragma unroll
    for (int ni = 0; ni < 2; ++ni) b[ni] = load_b_frag(BT, wn * 32 + ni * 16, r15, hsel);
#pragma unroll
    for (int mi = 0; mi < 2; ++mi)
#pragma unroll
      for (int ni = 0; ni < 2; ++ni)
        acc[mi][ni] = __builtin_amdgcn_wmma_f32_16x16x32_bf16(
            false, a[mi], false, b[ni], (short)0, acc[mi][ni], false, false);
    __syncthreads();
  }

  // epilogue: scale by routing coefficient, write/accumulate fp32 output
#pragma unroll
  for (int mi = 0; mi < 2; ++mi)
#pragma unroll
    for (int ni = 0; ni < 2; ++ni) {
      const int trow = t0 + wm * 32 + mi * 16 + hsel * 8;
      const int hcol = h0 + wn * 32 + ni * 16 + r15;
      const v8f av = acc[mi][ni];
#pragma unroll
      for (int r = 0; r < 8; ++r) {
        const int t = trow + r;
        const float cf = coef[(size_t)t * kE + eidx];
        const float v = cf * av[r];
        const size_t oi = (size_t)t * kH + hcol;
        out[oi] = accumulate ? (out[oi] + v) : v;
      }
    }
}

// ---------------------------------------------------------------------------
extern "C" void kernel_launch(void* const* d_in, const int* in_sizes, int n_in,
                              void* d_out, int out_size, void* d_ws, size_t ws_size,
                              hipStream_t stream) {
  (void)in_sizes; (void)n_in; (void)out_size; (void)ws_size;
  const float* x  = (const float*)d_in[0];   // [B,S,H]
  const float* wr = (const float*)d_in[1];   // [H,E]
  const float* wg = (const float*)d_in[2];   // [E,H,F]
  const float* wu = (const float*)d_in[3];   // [E,H,F]
  const float* wd = (const float*)d_in[4];   // [E,F,H]
  float* out = (float*)d_out;                // [T,H]

  // workspace: coef [T,E] fp32 (256 KB) | hid [T,F] bf16 (23 MB, reused per expert)
  float*  coef = (float*)d_ws;
  __bf16* hid  = (__bf16*)((char*)d_ws + (size_t)kT * kE * sizeof(float));

  router_topk_kernel<<<dim3(kT), dim3(128), 0, stream>>>(x, wr, coef);

  const dim3 blk(256);
  const dim3 grid_gu(kF / BN, kT / BM);   // 22 x 64
  const dim3 grid_dn(kH / BN, kT / BM);   // 32 x 64
  for (int e = 0; e < kE; ++e) {
    moe_gateup_wmma<<<grid_gu, blk, 0, stream>>>(
        x, wg + (size_t)e * kH * kF, wu + (size_t)e * kH * kF, coef, e, hid);
    moe_down_wmma<<<grid_dn, blk, 0, stream>>>(
        hid, wd + (size_t)e * kF * kH, coef, e, out, /*accumulate=*/(e != 0));
  }
}
